// ImprovedGNN_1443109011557
// MI455X (gfx1250) — matrix operand order repaired and checked by reading, not compiled, not run on hardware
//
#include <hip/hip_runtime.h>
#include <math.h>

// ---------------------------------------------------------------------------
// Types for CDNA5 WMMA (wave32): v_wmma_f32_16x16x32_bf16
// ---------------------------------------------------------------------------
typedef __attribute__((ext_vector_type(16))) __bf16 v16bf;
typedef __attribute__((ext_vector_type(8)))  __bf16 v8bf;
typedef __attribute__((ext_vector_type(8)))  float  v8f;

// async global->LDS pointer types: param1 = AS1 int4*, param2 = AS3 int4*
typedef int v4i __attribute__((ext_vector_type(4)));
typedef __attribute__((address_space(1))) v4i v4i_g;
typedef __attribute__((address_space(3))) v4i v4i_l;

#define BM 128
#define BN 64
#define BK 32
#define SAF 36   // LDS A stride (fp32 elems), padded vs 32 (144B rows, 16B aligned)
#define SB  40   // LDS B stride (bf16 elems), stored [n][k]

// Async global->LDS path (gfx1250): bypasses VGPRs, tracked by ASYNCcnt.
#if defined(__gfx1250__) && __has_builtin(__builtin_amdgcn_global_load_async_to_lds_b128)
#define HAVE_ASYNC_LDS 1
#else
#define HAVE_ASYNC_LDS 0
#endif

__device__ __forceinline__ void wait_asynccnt0() {
#if HAVE_ASYNC_LDS
#if __has_builtin(__builtin_amdgcn_s_wait_asynccnt)
    __builtin_amdgcn_s_wait_asynccnt(0);
#else
    asm volatile("s_wait_asynccnt 0x0" ::: "memory");
#endif
#endif
}

// ---------------------------------------------------------------------------
// GEMM: C[M x N] = A[M x K] @ B[K x N] (+ bias), fp32 in/out, bf16 WMMA core.
// Block: 256 threads = 8 waves. Block tile 128x64; wave tile 16x64 (4 WMMAs
// per K-step off one A fragment). Double-buffered LDS; A tile staged raw fp32
// via async-to-LDS, converted to bf16 at fragment build; B tile transposed and
// converted at staging (amortized across 8 consuming waves).
// K, N multiples of 32 (N multiple of 64); M arbitrary (guarded).
// ---------------------------------------------------------------------------
__global__ __launch_bounds__(256)
void gemm_bf16_wmma(const float* __restrict__ A, const float* __restrict__ B,
                    const float* __restrict__ bias, float* __restrict__ C,
                    int M, int K, int N)
{
    __shared__ __attribute__((aligned(16))) float  Asf[2][BM * SAF];
    __shared__ __attribute__((aligned(16))) __bf16 Bsb[2][BN * SB];

    const int t    = threadIdx.x;
    const int lane = t & 31;
    const int wave = t >> 5;
    const int m0   = blockIdx.x * BM;
    const int n0   = blockIdx.y * BN;
    const int half = lane >> 4;   // 0: lanes 0-15, 1: lanes 16-31
    const int mr   = lane & 15;

    // A staging: thread t owns 16 consecutive fp32 (half a 32-wide row)
    const int arow  = t >> 1;          // 0..127
    const int acol0 = (t & 1) * 16;    // 0 or 16
    const int gm_ld = m0 + arow;
    // B staging: thread t loads 8 consecutive fp32 of one k-row, stores transposed
    const int brow  = t >> 3;          // k: 0..31
    const int bn0   = (t & 7) * 8;     // n: 0..56

    v8f acc[4] = {{}, {}, {}, {}};

    const int nchunk = K / BK;

    auto stage = [&](int ci, int buf) {
        const int kk = ci * BK;
        // ---- A tile: raw fp32 into LDS ----
        if (gm_ld < M) {
            const float* gsrc = A + (size_t)gm_ld * K + kk + acol0;
            float*       ldst = &Asf[buf][arow * SAF + acol0];
#if HAVE_ASYNC_LDS
#pragma unroll
            for (int j = 0; j < 16; j += 4) {
                __builtin_amdgcn_global_load_async_to_lds_b128(
                    (v4i_g*)(gsrc + j), (v4i_l*)(ldst + j), 0, 0);
            }
#else
#pragma unroll
            for (int j = 0; j < 16; j += 4)
                *(float4*)(ldst + j) = *(const float4*)(gsrc + j);
#endif
        } else {
            float4 z = {0.0f, 0.0f, 0.0f, 0.0f};
            float* ldst = &Asf[buf][arow * SAF + acol0];
#pragma unroll
            for (int j = 0; j < 16; j += 4) *(float4*)(ldst + j) = z;
        }
        // ---- B tile: fp32 -> bf16, transposed to Bs[n][k] ----
        const float* gb = B + (size_t)(kk + brow) * N + n0 + bn0;
        const float4 fb0 = ((const float4*)gb)[0];
        const float4 fb1 = ((const float4*)gb)[1];
        __bf16* bd = &Bsb[buf][0];
        bd[(bn0 + 0) * SB + brow] = (__bf16)fb0.x;
        bd[(bn0 + 1) * SB + brow] = (__bf16)fb0.y;
        bd[(bn0 + 2) * SB + brow] = (__bf16)fb0.z;
        bd[(bn0 + 3) * SB + brow] = (__bf16)fb0.w;
        bd[(bn0 + 4) * SB + brow] = (__bf16)fb1.x;
        bd[(bn0 + 5) * SB + brow] = (__bf16)fb1.y;
        bd[(bn0 + 6) * SB + brow] = (__bf16)fb1.z;
        bd[(bn0 + 7) * SB + brow] = (__bf16)fb1.w;
    };

    stage(0, 0);
    wait_asynccnt0();
    __syncthreads();

    int cur = 0;
    for (int ci = 0; ci < nchunk; ++ci) {
        const int nxt = cur ^ 1;
        if (ci + 1 < nchunk) stage(ci + 1, nxt);   // overlap with compute below

        // A fragment (ISA 16-bit A 16x32): lane<16: m=lane, k in {0..7,16..23};
        // lane>=16: m=lane-16, k in {8..15,24..31}. Convert fp32->bf16 here.
        const float* ar = &Asf[cur][(wave * 16 + mr) * SAF];
        const float4 a0 = *(const float4*)(ar + half * 8);
        const float4 a1 = *(const float4*)(ar + half * 8 + 4);
        const float4 a2 = *(const float4*)(ar + 16 + half * 8);
        const float4 a3 = *(const float4*)(ar + 16 + half * 8 + 4);
        const v16bf afrag = {
            (__bf16)a0.x, (__bf16)a0.y, (__bf16)a0.z, (__bf16)a0.w,
            (__bf16)a1.x, (__bf16)a1.y, (__bf16)a1.z, (__bf16)a1.w,
            (__bf16)a2.x, (__bf16)a2.y, (__bf16)a2.z, (__bf16)a2.w,
            (__bf16)a3.x, (__bf16)a3.y, (__bf16)a3.z, (__bf16)a3.w };

        // B fragments (32x16): lane<16: n=lane, k in {0..7,16..23}; lane>=16: n=lane-16, k in {8..15,24..31}
        v16bf bfrag[4];
#pragma unroll
        for (int s = 0; s < 4; ++s) {
            const __bf16* br = &Bsb[cur][(s * 16 + mr) * SB + half * 8];
            v8bf lo = *(const v8bf*)(br);
            v8bf hi = *(const v8bf*)(br + 16);
            bfrag[s] = __builtin_shufflevector(lo, hi,
                         0,1,2,3,4,5,6,7,8,9,10,11,12,13,14,15);
        }

#pragma unroll
        for (int s = 0; s < 4; ++s)
            acc[s] = __builtin_amdgcn_wmma_f32_16x16x32_bf16(
                         false, afrag, false, bfrag[s], (short)0, acc[s], false, false);

        wait_asynccnt0();   // next tile's async A copies landed
        __syncthreads();    // everyone done reading cur / writing nxt
        cur = nxt;
    }

    // C/D layout: VGPR r -> row (r + half*8), col = lane&15 (per 16-col subtile)
    const int gn = n0 + mr;
    float badd[4] = {0.0f, 0.0f, 0.0f, 0.0f};
    if (bias) {
#pragma unroll
        for (int s = 0; s < 4; ++s) badd[s] = bias[gn + 16 * s];
    }
#pragma unroll
    for (int r = 0; r < 8; ++r) {
        const int gm = m0 + wave * 16 + half * 8 + r;
        if (gm < M) {
#pragma unroll
            for (int s = 0; s < 4; ++s)
                C[(size_t)gm * N + gn + 16 * s] = acc[s][r] + badd[s];
        }
    }
}

// ---------------------------------------------------------------------------
// Wave32 butterfly sum
// ---------------------------------------------------------------------------
__device__ __forceinline__ float wave_sum(float v) {
#pragma unroll
    for (int m = 16; m > 0; m >>= 1) v += __shfl_xor(v, m, 32);
    return v;
}

// ---------------------------------------------------------------------------
// LayerNorm over H=256 (+optional pre-bias, ReLU, residual). One wave per row.
// ---------------------------------------------------------------------------
__global__ __launch_bounds__(256)
void ln_kernel(const float* __restrict__ in, const float* __restrict__ prebias,
               const float* __restrict__ gamma, const float* __restrict__ beta,
               const float* __restrict__ residual, float* __restrict__ out,
               int rows, int relu)
{
    const int lane = threadIdx.x & 31;
    const int row  = blockIdx.x * 8 + (threadIdx.x >> 5);
    if (row >= rows) return;

    const float* x = in + (size_t)row * 256;
    float v[8];
    float s = 0.0f;
#pragma unroll
    for (int j = 0; j < 8; ++j) {
        const int f = lane + j * 32;
        float t = x[f];
        if (prebias) t += prebias[f];
        v[j] = t; s += t;
    }
    const float mean = wave_sum(s) * (1.0f / 256.0f);
    float vs = 0.0f;
#pragma unroll
    for (int j = 0; j < 8; ++j) { const float d = v[j] - mean; vs += d * d; }
    const float var = wave_sum(vs) * (1.0f / 256.0f);
    const float rs  = rsqrtf(var + 1e-5f);
#pragma unroll
    for (int j = 0; j < 8; ++j) {
        const int f = lane + j * 32;
        float y = (v[j] - mean) * rs * gamma[f] + beta[f];
        if (relu) y = fmaxf(y, 0.0f);
        if (residual) y += residual[(size_t)row * 256 + f];
        out[(size_t)row * 256 + f] = y;
    }
}

// ---------------------------------------------------------------------------
// Small utility kernels
// ---------------------------------------------------------------------------
__global__ __launch_bounds__(256)
void fill_kernel(float* p, int n, float val) {
    const int i = blockIdx.x * 256 + threadIdx.x;
    if (i < n) p[i] = val;
}

__global__ __launch_bounds__(256)
void degcount_kernel(float* deg, const int* __restrict__ dst, int ne) {
    const int e = blockIdx.x * 256 + threadIdx.x;
    if (e < ne) atomicAdd(&deg[dst[e]], 1.0f);
}

__global__ __launch_bounds__(256)
void rsqrt_kernel(float* p, int n) {
    const int i = blockIdx.x * 256 + threadIdx.x;
    if (i < n) p[i] = rsqrtf(p[i]);   // deg >= 1 always (self loop)
}

// agg[i] = tmp[i] * dinv[i]^2   (self loop term), H=256
__global__ __launch_bounds__(256)
void selfloop_kernel(const float* __restrict__ tmp, const float* __restrict__ dinv,
                     float* __restrict__ agg, int n)
{
    const int tid = blockIdx.x * 256 + threadIdx.x;
    const int i   = tid >> 6;
    const int f0  = (tid & 63) * 4;
    if (i >= n) return;
    const float w = dinv[i] * dinv[i];
    float4 v = *(const float4*)&tmp[(size_t)i * 256 + f0];
    v.x *= w; v.y *= w; v.z *= w; v.w *= w;
    *(float4*)&agg[(size_t)i * 256 + f0] = v;
}

// agg[dst] += tmp[src] * dinv[src]*dinv[dst], H=256, 64 threads/edge
__global__ __launch_bounds__(256)
void edge_scatter_kernel(const float* __restrict__ tmp, const float* __restrict__ dinv,
                         const int* __restrict__ src, const int* __restrict__ dst,
                         float* __restrict__ agg, int ne)
{
    const int tid = blockIdx.x * 256 + threadIdx.x;
    const int e   = tid >> 6;
    const int f0  = (tid & 63) * 4;
    if (e >= ne) return;
    const int s = src[e], d = dst[e];
    const float w = dinv[s] * dinv[d];
    const float4 v = *(const float4*)&tmp[(size_t)s * 256 + f0];
    float* o = &agg[(size_t)d * 256 + f0];
    atomicAdd(o + 0, v.x * w);
    atomicAdd(o + 1, v.y * w);
    atomicAdd(o + 2, v.z * w);
    atomicAdd(o + 3, v.w * w);
}

// a[i] = tanh(dot(x3[i,:], attn_w) + attn_b). One wave per node row.
__global__ __launch_bounds__(256)
void attn_kernel(const float* __restrict__ x3, const float* __restrict__ aw,
                 const float* __restrict__ ab, float* __restrict__ alog, int n)
{
    const int lane = threadIdx.x & 31;
    const int i    = blockIdx.x * 8 + (threadIdx.x >> 5);
    if (i >= n) return;
    float s = 0.0f;
#pragma unroll
    for (int j = 0; j < 8; ++j) {
        const int f = lane + j * 32;
        s += x3[(size_t)i * 256 + f] * aw[f];
    }
    s = wave_sum(s);
    if (lane == 0) alog[i] = tanhf(s + ab[0]);
}

// mode 0: block-partial max of a; mode 1: block-partial sum of exp(a - scal[0])
__global__ __launch_bounds__(256)
void part_reduce(const float* __restrict__ a, const float* __restrict__ scal,
                 float* __restrict__ part, int n, int mode)
{
    __shared__ float sm[256];
    const int t = threadIdx.x;
    const float gm = mode ? scal[0] : 0.0f;
    float acc = mode ? 0.0f : -3.4e38f;
    for (int i = blockIdx.x * 256 + t; i < n; i += 256 * gridDim.x) {
        const float x = a[i];
        if (mode) acc += expf(x - gm); else acc = fmaxf(acc, x);
    }
    sm[t] = acc;
    __syncthreads();
    for (int s = 128; s > 0; s >>= 1) {
        if (t < s) sm[t] = mode ? (sm[t] + sm[t + s]) : fmaxf(sm[t], sm[t + s]);
        __syncthreads();
    }
    if (t == 0) part[blockIdx.x] = sm[0];
}

__global__ __launch_bounds__(256)
void final_reduce(const float* __restrict__ part, float* __restrict__ scal,
                  int nparts, int mode)
{
    __shared__ float sm[256];
    const int t = threadIdx.x;
    float acc = mode ? 0.0f : -3.4e38f;
    for (int i = t; i < nparts; i += 256) {
        const float x = part[i];
        if (mode) acc += x; else acc = fmaxf(acc, x);
    }
    sm[t] = acc;
    __syncthreads();
    for (int s = 128; s > 0; s >>= 1) {
        if (t < s) sm[t] = mode ? (sm[t] + sm[t + s]) : fmaxf(sm[t], sm[t + s]);
        __syncthreads();
    }
    if (t == 0) scal[mode ? 1 : 0] = sm[0];
}

// pooled[batch[i]] += x3[i] * softmax(a)[i], H=256, 64 threads/node
__global__ __launch_bounds__(256)
void pool_kernel(const float* __restrict__ x3, const float* __restrict__ alog,
                 const float* __restrict__ scal, const int* __restrict__ batch,
                 float* __restrict__ pooled, int n)
{
    const int tid = blockIdx.x * 256 + threadIdx.x;
    const int i   = tid >> 6;
    const int f0  = (tid & 63) * 4;
    if (i >= n) return;
    const float coef = expf(alog[i] - scal[0]) / scal[1];
    const int g = batch[i];
    const float4 v = *(const float4*)&x3[(size_t)i * 256 + f0];
    float* o = &pooled[(size_t)g * 256 + f0];
    atomicAdd(o + 0, v.x * coef);
    atomicAdd(o + 1, v.y * coef);
    atomicAdd(o + 2, v.z * coef);
    atomicAdd(o + 3, v.w * coef);
}

// L2 row-normalize [rows x 128] in place. One wave per row, 4 floats/lane.
__global__ __launch_bounds__(256)
void l2norm_kernel(float* __restrict__ out, int rows)
{
    const int lane = threadIdx.x & 31;
    const int row  = blockIdx.x * 8 + (threadIdx.x >> 5);
    if (row >= rows) return;
    float4 v = *(const float4*)&out[(size_t)row * 128 + lane * 4];
    float ss = v.x * v.x + v.y * v.y + v.z * v.z + v.w * v.w;
    ss = wave_sum(ss);
    const float inv = 1.0f / fmaxf(sqrtf(ss), 1e-12f);
    v.x *= inv; v.y *= inv; v.z *= inv; v.w *= inv;
    *(float4*)&out[(size_t)row * 128 + lane * 4] = v;
}

// ---------------------------------------------------------------------------
// Launcher
// ---------------------------------------------------------------------------
extern "C" void kernel_launch(void* const* d_in, const int* in_sizes, int n_in,
                              void* d_out, int out_size, void* d_ws, size_t ws_size,
                              hipStream_t stream)
{
    const float* x      = (const float*)d_in[0];
    const int*   ei     = (const int*)d_in[1];
    const int*   batch  = (const int*)d_in[2];
    const float* enc_w  = (const float*)d_in[3];
    const float* enc_b  = (const float*)d_in[4];
    const float* enc_g  = (const float*)d_in[5];
    const float* enc_be = (const float*)d_in[6];
    const float* w1 = (const float*)d_in[7],  *b1 = (const float*)d_in[8];
    const float* g1 = (const float*)d_in[9],  *be1 = (const float*)d_in[10];
    const float* w2 = (const float*)d_in[11], *b2 = (const float*)d_in[12];
    const float* g2 = (const float*)d_in[13], *be2 = (const float*)d_in[14];
    const float* w3 = (const float*)d_in[15], *b3 = (const float*)d_in[16];
    const float* g3 = (const float*)d_in[17], *be3 = (const float*)d_in[18];
    const float* attn_w = (const float*)d_in[19];
    const float* attn_b = (const float*)d_in[20];
    const float* pw1 = (const float*)d_in[21], *pb1 = (const float*)d_in[22];
    const float* pg  = (const float*)d_in[23], *pbe = (const float*)d_in[24];
    const float* pw2 = (const float*)d_in[25], *pb2 = (const float*)d_in[26];

    const int H    = in_sizes[4];          // 256
    const int F    = in_sizes[3] / H;      // 128
    const int N    = in_sizes[0] / F;      // 20000
    const int NE   = in_sizes[1] / 2;      // 320000
    const int EDIM = in_sizes[26];         // 128
    const int G    = out_size / EDIM;      // 64
    const int* srcp = ei;
    const int* dstp = ei + NE;

    // --- workspace carve (all fp32) ---
    char* wp = (char*)d_ws;
    auto carve = [&](size_t bytes) -> float* {
        float* p = (float*)wp;
        wp += (bytes + 255) & ~(size_t)255;
        return p;
    };
    float* buf0   = carve((size_t)N * H * 4);
    float* buf1   = carve((size_t)N * H * 4);
    float* buf2   = carve((size_t)N * H * 4);
    float* deg    = carve((size_t)N * 4);       // becomes dinv
    float* alog   = carve((size_t)N * 4);
    float* red    = carve(512 * 4);
    float* scal   = carve(16 * 4);
    float* pooled = carve((size_t)G * H * 4);
    float* pA     = carve((size_t)G * H * 4);
    float* pB     = carve((size_t)G * H * 4);
    (void)n_in; (void)ws_size;

    auto gemm = [&](const float* A, const float* B, const float* bias, float* C,
                    int M, int K, int Nc) {
        dim3 grid((M + BM - 1) / BM, Nc / BN);
        gemm_bf16_wmma<<<grid, 256, 0, stream>>>(A, B, bias, C, M, K, Nc);
    };
    auto ln = [&](const float* in, const float* prebias, const float* g,
                  const float* be, const float* res, float* out, int rows, int relu) {
        ln_kernel<<<(rows + 7) / 8, 256, 0, stream>>>(in, prebias, g, be, res, out, rows, relu);
    };

    const int nb_N   = (N + 255) / 256;
    const int nb_NE  = (NE + 255) / 256;
    const int nb_NH4 = ((N * 64) + 255) / 256;   // N*H/4 threads
    const int nb_EH4 = ((NE * 64) + 255) / 256;  // NE*H/4 threads

    // 1) degrees -> dinv (computed once, shared by all three GCN layers)
    fill_kernel<<<nb_N, 256, 0, stream>>>(deg, N, 1.0f);
    degcount_kernel<<<nb_NE, 256, 0, stream>>>(deg, dstp, NE);
    rsqrt_kernel<<<nb_N, 256, 0, stream>>>(deg, N);

    // 2) encoder: h = relu(LN(x @ enc_w + enc_b))
    gemm(x, enc_w, enc_b, buf1, N, F, H);
    ln(buf1, nullptr, enc_g, enc_be, nullptr, buf0, N, 1);        // buf0 = h

    // 3) layer 1: x1 = relu(LN(gcn(h)))
    gemm(buf0, w1, nullptr, buf1, N, H, H);                       // buf1 = h@W1
    selfloop_kernel<<<nb_NH4, 256, 0, stream>>>(buf1, deg, buf2, N);
    edge_scatter_kernel<<<nb_EH4, 256, 0, stream>>>(buf1, deg, srcp, dstp, buf2, NE);
    ln(buf2, b1, g1, be1, nullptr, buf1, N, 1);                   // buf1 = x1

    // 4) layer 2: x2 = relu(LN(gcn(x1))) + x1
    gemm(buf1, w2, nullptr, buf0, N, H, H);                       // buf0 = x1@W2
    selfloop_kernel<<<nb_NH4, 256, 0, stream>>>(buf0, deg, buf2, N);
    edge_scatter_kernel<<<nb_EH4, 256, 0, stream>>>(buf0, deg, srcp, dstp, buf2, NE);
    ln(buf2, b2, g2, be2, buf1, buf0, N, 1);                      // buf0 = x2

    // 5) layer 3: x3 = relu(LN(gcn(x2))) + x2
    gemm(buf0, w3, nullptr, buf1, N, H, H);                       // buf1 = x2@W3
    selfloop_kernel<<<nb_NH4, 256, 0, stream>>>(buf1, deg, buf2, N);
    edge_scatter_kernel<<<nb_EH4, 256, 0, stream>>>(buf1, deg, srcp, dstp, buf2, NE);
    ln(buf2, b3, g3, be3, buf0, buf1, N, 1);                      // buf1 = x3

    // 6) attention logits + global softmax over nodes
    attn_kernel<<<(N + 7) / 8, 256, 0, stream>>>(buf1, attn_w, attn_b, alog, N);
    part_reduce<<<256, 256, 0, stream>>>(alog, scal, red, N, 0);
    final_reduce<<<1, 256, 0, stream>>>(red, scal, 256, 0);       // scal[0] = max
    part_reduce<<<256, 256, 0, stream>>>(alog, scal, red, N, 1);
    final_reduce<<<1, 256, 0, stream>>>(red, scal, 256, 1);       // scal[1] = sum

    // 7) attention-weighted pooling into [G, H]
    fill_kernel<<<(G * H + 255) / 256, 256, 0, stream>>>(pooled, G * H, 0.0f);
    pool_kernel<<<nb_NH4, 256, 0, stream>>>(buf1, alog, scal, batch, pooled, N);

    // 8) graph head: relu(LN(pooled @ pw1 + pb1)) @ pw2 + pb2, then L2 normalize
    gemm(pooled, pw1, pb1, pB, G, H, H);
    ln(pB, nullptr, pg, pbe, nullptr, pA, G, 1);
    gemm(pA, pw2, pb2, (float*)d_out, G, H, EDIM);
    l2norm_kernel<<<(G + 7) / 8, 256, 0, stream>>>((float*)d_out, G);
}
